// Decoder_84232898609865
// MI455X (gfx1250) — matrix operand order repaired and checked by reading, not compiled, not run on hardware
//
#include <hip/hip_runtime.h>
#include <hip/hip_bf16.h>
#include <math.h>

// Problem constants (match reference)
#define NN    20000     // nodes
#define NE    160000    // edges
#define HIDD  128
#define FINN  64
#define SKIPC 2

typedef __attribute__((ext_vector_type(16))) __bf16 v16bf;
typedef __attribute__((ext_vector_type(8)))  __bf16 v8bf;
typedef __attribute__((ext_vector_type(8)))  float  v8f;
typedef __attribute__((ext_vector_type(4)))  unsigned int u32x4;
typedef __attribute__((ext_vector_type(8)))  int i32x8;
typedef __attribute__((ext_vector_type(4)))  int i32x4;

#define HAVE_TDM __has_builtin(__builtin_amdgcn_tensor_load_to_lds)

#if __has_builtin(__builtin_amdgcn_s_wait_tensorcnt)
#define WAIT_TENSORCNT(n) __builtin_amdgcn_s_wait_tensorcnt(n)
#else
#define WAIT_TENSORCNT(n) asm volatile("s_wait_tensorcnt %0" ::"i"(n) : "memory")
#endif

// ---------------------------------------------------------------------------
// helpers
// ---------------------------------------------------------------------------
__device__ __forceinline__ __bf16 f2bf(float f) {
  unsigned u = __float_as_uint(f);
  unsigned r = u + 0x7FFFu + ((u >> 16) & 1u);   // round-to-nearest-even
  unsigned short h = (unsigned short)(r >> 16);
  return __builtin_bit_cast(__bf16, h);
}

__device__ __forceinline__ float sigm(float x) { return 1.0f / (1.0f + __expf(-x)); }

__device__ __forceinline__ void atomicMaxFloat(float* addr, float val) {
  int iv = __float_as_int(val);
  if (iv >= 0) atomicMax((int*)addr, iv);
  else         atomicMin((unsigned int*)addr, (unsigned int)iv);
}

__device__ __forceinline__ float waveReduceSum(float v) {
  #pragma unroll
  for (int off = 16; off > 0; off >>= 1) v += __shfl_xor(v, off, 32);
  return v;
}

// ---------------------------------------------------------------------------
// TDM tile load: 2D tile (32 k-elements x 64 rows) of bf16 from a row-major
// [rows x Kp] matrix into LDS with 80B row stride (64B data + 16B pad).
//   data_size=1 (2 bytes); pad_interval=3 (16 DW = 64B); pad_amount=3 (4 DW = 16B)
//   tensor_dim0=Kp, tensor_dim1=rows  (OOB rows are hardware zero-filled)
// ---------------------------------------------------------------------------
#if HAVE_TDM
__device__ __forceinline__ void tdm_load_tile(const __bf16* gbase, int row0, int k0,
                                              int rows, int Kp, unsigned ldsOff)
{
  unsigned long long ga =
      (unsigned long long)(size_t)(gbase + (size_t)row0 * Kp + k0);
  u32x4 g0;
  g0[0] = 1u;                                            // count=1, user descriptor
  g0[1] = ldsOff;                                        // lds_addr (bytes)
  g0[2] = (unsigned)(ga & 0xFFFFFFFFu);                  // global_addr[31:0]
  g0[3] = (unsigned)((ga >> 32) & 0x1FFFFFFu) | (2u << 30); // addr[56:32] | type=2
  i32x8 g1;
  g1[0] = (int)((1u << 16)      // data_size = 2 bytes
              | (1u << 20)      // pad_enable
              | (3u << 22)      // pad_interval: 16 DWORDs (64B)
              | (3u << 25));    // pad_amount: 4 DWORDs (16B)
  g1[1] = (int)(((unsigned)Kp & 0xFFFFu) << 16);                 // tensor_dim0 lo
  g1[2] = (int)((((unsigned)Kp >> 16) & 0xFFFFu) |
                (((unsigned)rows & 0xFFFFu) << 16));             // dim0 hi | dim1 lo
  g1[3] = (int)((((unsigned)rows >> 16) & 0xFFFFu) | (32u << 16)); // dim1 hi | tile_dim0=32
  g1[4] = 64;                                                    // tile_dim1=64, tile_dim2=0
  g1[5] = Kp;                                                    // tensor_dim0_stride lo
  g1[6] = 0;
  g1[7] = 0;
  i32x4 gz4 = {};
  i32x8 gz8 = {};
  // 6-arg toolchain form: (g0, g1, g2, g3, extra, cpol)
  __builtin_amdgcn_tensor_load_to_lds(g0, g1, gz4, gz4, gz8, 0);
}
#endif

// ---------------------------------------------------------------------------
// fragment math: A frag K map = 8*half + i (i<8) / 16+8*half+(i-8);
//                B frag K map = same as A ; both = two 16B LDS loads
// ---------------------------------------------------------------------------
__device__ __forceinline__ void compute_step(const __bf16 (*Asb)[40], const __bf16 (*Bsb)[40],
                                             int rt, int cp, int half, int l16,
                                             v8f& acc0, v8f& acc1)
{
  const int arow = rt * 16 + l16;
  v8bf alo = *(const v8bf*)&Asb[arow][half * 8];
  v8bf ahi = *(const v8bf*)&Asb[arow][16 + half * 8];
  v16bf afrag;
  #pragma unroll
  for (int i = 0; i < 8; ++i) { afrag[i] = alo[i]; afrag[i + 8] = ahi[i]; }

  #pragma unroll
  for (int cb = 0; cb < 2; ++cb) {
    int n = cp * 32 + cb * 16 + l16;
    v8bf blo = *(const v8bf*)&Bsb[n][half * 8];
    v8bf bhi = *(const v8bf*)&Bsb[n][16 + half * 8];
    v16bf bfrag;
    #pragma unroll
    for (int i = 0; i < 8; ++i) { bfrag[i] = blo[i]; bfrag[i + 8] = bhi[i]; }
    if (cb == 0)
      acc0 = __builtin_amdgcn_wmma_f32_16x16x32_bf16(false, afrag, false, bfrag,
                                                     (short)0, acc0, false, false);
    else
      acc1 = __builtin_amdgcn_wmma_f32_16x16x32_bf16(false, afrag, false, bfrag,
                                                     (short)0, acc1, false, false);
  }
}

// ---------------------------------------------------------------------------
// WMMA GEMM:  C[M,Nn] = A[M,Kp] @ Bt[Nn,Kp]^T + bias
// A, Bt are bf16, K pre-padded to a multiple of 32 (zero tail), Nn % 64 == 0.
// Tile 64x64x32; 256 threads = 8 waves; TDM-staged double-buffered LDS.
// ---------------------------------------------------------------------------
__global__ __launch_bounds__(256)
void wmma_gemm_bias(const __bf16* __restrict__ A, const __bf16* __restrict__ Bt,
                    const float* __restrict__ bias, float* __restrict__ Cmat,
                    int M, int Nn, int Kp)
{
  __shared__ __align__(16) __bf16 As[2][64][40];   // 80B row stride (TDM pad)
  __shared__ __align__(16) __bf16 Bs[2][64][40];

  const int t    = threadIdx.x;
  const int lane = t & 31;
  const int wave = t >> 5;
  const int rowBase = blockIdx.y * 64;
  const int colBase = blockIdx.x * 64;
  const int rt   = wave & 3;
  const int cp   = wave >> 2;
  const int half = lane >> 4;
  const int l16  = lane & 15;
  const int nk   = Kp >> 5;

  v8f acc0 = {}; v8f acc1 = {};

#if HAVE_TDM
  const unsigned asOff0 = (unsigned)(size_t)&As[0][0][0];
  const unsigned asOff1 = (unsigned)(size_t)&As[1][0][0];
  const unsigned bsOff0 = (unsigned)(size_t)&Bs[0][0][0];
  const unsigned bsOff1 = (unsigned)(size_t)&Bs[1][0][0];

  if (t < 32) {   // wave 0 drives the Tensor Data Mover (EXEC ignored by TDM)
    tdm_load_tile(A,  rowBase, 0, M,  Kp, asOff0);
    tdm_load_tile(Bt, colBase, 0, Nn, Kp, bsOff0);
  }
  for (int kt = 0; kt < nk; ++kt) {
    const int cur = kt & 1;
    if (t < 32) {
      if (kt + 1 < nk) {   // prefetch next tile into the other buffer
        tdm_load_tile(A,  rowBase, (kt + 1) << 5, M,  Kp, cur ? asOff0 : asOff1);
        tdm_load_tile(Bt, colBase, (kt + 1) << 5, Nn, Kp, cur ? bsOff0 : bsOff1);
        WAIT_TENSORCNT(2);   // tile kt landed (TDM ops complete in order)
      } else {
        WAIT_TENSORCNT(0);
      }
    }
    __syncthreads();
    compute_step(As[cur], Bs[cur], rt, cp, half, l16, acc0, acc1);
    __syncthreads();   // all waves done reading before this buffer is re-filled
  }
#else
  for (int kt = 0; kt < nk; ++kt) {
    const int k0 = kt << 5;
    __syncthreads();
    #pragma unroll
    for (int i = 0; i < 8; ++i) {
      int idx = t + i * 256;
      int r = idx >> 5, c = idx & 31;
      int gr = rowBase + r;
      As[0][r][c] = (gr < M) ? A[(size_t)gr * Kp + k0 + c]
                             : __builtin_bit_cast(__bf16, (unsigned short)0);
      int gn = colBase + r;
      Bs[0][r][c] = (gn < Nn) ? Bt[(size_t)gn * Kp + k0 + c]
                              : __builtin_bit_cast(__bf16, (unsigned short)0);
    }
    __syncthreads();
    compute_step(As[0], Bs[0], rt, cp, half, l16, acc0, acc1);
  }
#endif

  // C/D layout: VGPR r -> row = 8*half + r; col = lane&15
  const int col0 = colBase + cp * 32 + l16;
  const int col1 = col0 + 16;
  #pragma unroll
  for (int r = 0; r < 8; ++r) {
    int row = rowBase + rt * 16 + half * 8 + r;
    if (row < M) {
      Cmat[(size_t)row * Nn + col0] = acc0[r] + (bias ? bias[col0] : 0.0f);
      Cmat[(size_t)row * Nn + col1] = acc1[r] + (bias ? bias[col1] : 0.0f);
    }
  }
}

// ---------------------------------------------------------------------------
// Prep kernels: bf16 A construction (concat, K zero-padded) + weight transpose
// ---------------------------------------------------------------------------
__global__ void concat2_bf16_kernel(const float* __restrict__ a, int ca,
                                    const float* __restrict__ b, int cb,
                                    __bf16* __restrict__ out, int rows, int Kp) {
  int i = blockIdx.x * blockDim.x + threadIdx.x;
  if (i >= rows * Kp) return;
  int r = i / Kp, c = i - r * Kp;
  float v = 0.0f;
  if (c < ca)           v = a[(size_t)r * ca + c];
  else if (c < ca + cb) v = b[(size_t)r * cb + (c - ca)];
  out[i] = f2bf(v);
}

// Bt[n][k] = W[k][n] (bf16), k zero-padded to Kp
__global__ void transpose_cast_kernel(const float* __restrict__ W,
                                      __bf16* __restrict__ Bt, int K, int Kp, int Nn) {
  int i = blockIdx.x * blockDim.x + threadIdx.x;
  if (i >= Nn * Kp) return;
  int n = i / Kp, k = i - n * Kp;
  Bt[i] = f2bf(k < K ? W[(size_t)k * Nn + n] : 0.0f);
}

// ---------------------------------------------------------------------------
// Elementwise / utility kernels
// ---------------------------------------------------------------------------
__global__ void fill_kernel(float* p, float v, int n) {
  int i = blockIdx.x * blockDim.x + threadIdx.x;
  if (i < n) p[i] = v;
}

__global__ void relu_kernel(float* p, int n) {
  int i = blockIdx.x * blockDim.x + threadIdx.x;
  if (i < n) p[i] = fmaxf(p[i], 0.0f);
}

__global__ void sigmoid_store_kernel(const float* __restrict__ x, float* __restrict__ y, int n) {
  int i = blockIdx.x * blockDim.x + threadIdx.x;
  if (i < n) y[i] = sigm(x[i]);
}

// ---------------------------------------------------------------------------
// Edge phase (TransformerConv softmax-by-destination), wave-per-edge
// ---------------------------------------------------------------------------
__global__ void edge_logits_kernel(const float* __restrict__ q, const float* __restrict__ k,
                                   const float* __restrict__ ew, const float* __restrict__ We,
                                   const int* __restrict__ src, const int* __restrict__ dst,
                                   float* __restrict__ logits, float* __restrict__ mbuf,
                                   int E, int O, float scale)
{
  int e    = (blockIdx.x * blockDim.x + threadIdx.x) >> 5;
  int lane = threadIdx.x & 31;
  if (e >= E) return;
  int s = src[e], d = dst[e];
  float w0 = ew[(size_t)e * 2 + 0], w1 = ew[(size_t)e * 2 + 1];
  float acc = 0.0f;
  for (int j = lane; j < O; j += 32) {
    float ee = w0 * We[j] + w1 * We[O + j];
    acc += q[(size_t)d * O + j] * (k[(size_t)s * O + j] + ee);
  }
  acc = waveReduceSum(acc);
  if (lane == 0) {
    float lg = acc * scale;
    logits[e] = lg;
    atomicMaxFloat(&mbuf[d], lg);
  }
}

__global__ void edge_exp_kernel(const float* __restrict__ logits, const float* __restrict__ mbuf,
                                const int* __restrict__ dst,
                                float* __restrict__ pexp, float* __restrict__ ssum, int E) {
  int e = blockIdx.x * blockDim.x + threadIdx.x;
  if (e >= E) return;
  int d = dst[e];
  float p = __expf(logits[e] - mbuf[d]);
  pexp[e] = p;
  atomicAdd(&ssum[d], p);
}

__global__ void edge_scatter_kernel(const float* __restrict__ v, const float* __restrict__ ew,
                                    const float* __restrict__ We,
                                    const int* __restrict__ src, const int* __restrict__ dst,
                                    const float* __restrict__ pexp, const float* __restrict__ ssum,
                                    float* __restrict__ out, int E, int O)
{
  int e    = (blockIdx.x * blockDim.x + threadIdx.x) >> 5;
  int lane = threadIdx.x & 31;
  if (e >= E) return;
  int s = src[e], d = dst[e];
  float w0 = ew[(size_t)e * 2 + 0], w1 = ew[(size_t)e * 2 + 1];
  float alpha = pexp[e] / ssum[d];
  for (int j = lane; j < O; j += 32) {
    float ee = w0 * We[j] + w1 * We[O + j];
    atomicAdd(&out[(size_t)d * O + j], (v[(size_t)s * O + j] + ee) * alpha);
  }
}

// ---------------------------------------------------------------------------
// LSTM gate + LayerNorm
// ---------------------------------------------------------------------------
__global__ void lstm_gate_kernel(const float* __restrict__ z, const float* __restrict__ cprev,
                                 float* __restrict__ h2, float* __restrict__ c2, int n) {
  int idx = blockIdx.x * blockDim.x + threadIdx.x;
  if (idx >= n * HIDD) return;
  int node = idx >> 7, d = idx & 127;
  const float* zr = z + (size_t)node * (4 * HIDD);
  float gi = sigm(zr[d]);
  float gf = sigm(zr[HIDD + d]);
  float gg = tanhf(zr[2 * HIDD + d]);
  float go = sigm(zr[3 * HIDD + d]);
  float c  = gf * cprev[idx] + gi * gg;
  h2[idx] = go * tanhf(c);
  c2[idx] = c;
}

__global__ void layernorm_kernel(const float* __restrict__ x, const float* __restrict__ w,
                                 const float* __restrict__ b, float* __restrict__ y,
                                 int rows, int doRelu)
{
  int row  = (blockIdx.x * blockDim.x + threadIdx.x) >> 5;
  int lane = threadIdx.x & 31;
  if (row >= rows) return;
  const float* xr = x + (size_t)row * HIDD;
  float v[4];
  float s = 0.0f;
  #pragma unroll
  for (int i = 0; i < 4; ++i) { v[i] = xr[lane + 32 * i]; s += v[i]; }
  s = waveReduceSum(s);
  float mean = s * (1.0f / HIDD);
  float var = 0.0f;
  #pragma unroll
  for (int i = 0; i < 4; ++i) { float d = v[i] - mean; var += d * d; }
  var = waveReduceSum(var) * (1.0f / HIDD);
  float inv = rsqrtf(var + 1e-5f);
  #pragma unroll
  for (int i = 0; i < 4; ++i) {
    int j = lane + 32 * i;
    float o = (v[i] - mean) * inv * w[j] + b[j];
    if (doRelu) o = fmaxf(o, 0.0f);
    y[(size_t)row * HIDD + j] = o;
  }
}

// ---------------------------------------------------------------------------
// fc2 head (O = 1): wave-per-node quad dot product, then scalar edge phase
// ---------------------------------------------------------------------------
__global__ void fc2_proj_kernel(const float* __restrict__ x,
                                const float* __restrict__ Wq, const float* __restrict__ bq,
                                const float* __restrict__ Wk, const float* __restrict__ bk,
                                const float* __restrict__ Wv, const float* __restrict__ bv,
                                const float* __restrict__ Ws, const float* __restrict__ bs,
                                float* __restrict__ q1, float* __restrict__ k1,
                                float* __restrict__ v1, float* __restrict__ o1, int rows)
{
  int row  = (blockIdx.x * blockDim.x + threadIdx.x) >> 5;
  int lane = threadIdx.x & 31;
  if (row >= rows) return;
  float aq = 0, ak = 0, av = 0, as_ = 0;
  #pragma unroll
  for (int i = 0; i < 4; ++i) {
    int j = lane + 32 * i;
    float xv = x[(size_t)row * HIDD + j];
    aq += xv * Wq[j]; ak += xv * Wk[j]; av += xv * Wv[j]; as_ += xv * Ws[j];
  }
  aq = waveReduceSum(aq); ak = waveReduceSum(ak);
  av = waveReduceSum(av); as_ = waveReduceSum(as_);
  if (lane == 0) {
    q1[row] = aq + bq[0]; k1[row] = ak + bk[0];
    v1[row] = av + bv[0]; o1[row] = as_ + bs[0];
  }
}

__global__ void fc2_edge_logits_kernel(const float* __restrict__ q1, const float* __restrict__ k1,
                                       const float* __restrict__ ew, const float* __restrict__ We,
                                       const int* __restrict__ src, const int* __restrict__ dst,
                                       float* __restrict__ logits, float* __restrict__ mbuf, int E) {
  int e = blockIdx.x * blockDim.x + threadIdx.x;
  if (e >= E) return;
  int s = src[e], d = dst[e];
  float ee = ew[(size_t)e * 2 + 0] * We[0] + ew[(size_t)e * 2 + 1] * We[1];
  float lg = q1[d] * (k1[s] + ee);
  logits[e] = lg;
  atomicMaxFloat(&mbuf[d], lg);
}

__global__ void fc2_scatter_kernel(const float* __restrict__ v1, const float* __restrict__ ew,
                                   const float* __restrict__ We,
                                   const int* __restrict__ src, const int* __restrict__ dst,
                                   const float* __restrict__ pexp, const float* __restrict__ ssum,
                                   float* __restrict__ out, int E) {
  int e = blockIdx.x * blockDim.x + threadIdx.x;
  if (e >= E) return;
  int s = src[e], d = dst[e];
  float ee = ew[(size_t)e * 2 + 0] * We[0] + ew[(size_t)e * 2 + 1] * We[1];
  float alpha = pexp[e] / ssum[d];
  atomicAdd(&out[d], (v1[s] + ee) * alpha);
}

// ---------------------------------------------------------------------------
// Host orchestration
// ---------------------------------------------------------------------------
struct TP { const float *Wq,*bq,*Wk,*bk,*Wv,*bv,*We,*Ws,*bs; };
static TP getTP(void* const* d_in, int base) {
  TP p;
  p.Wq = (const float*)d_in[base + 0]; p.bq = (const float*)d_in[base + 1];
  p.Wk = (const float*)d_in[base + 2]; p.bk = (const float*)d_in[base + 3];
  p.Wv = (const float*)d_in[base + 4]; p.bv = (const float*)d_in[base + 5];
  p.We = (const float*)d_in[base + 6];
  p.Ws = (const float*)d_in[base + 7]; p.bs = (const float*)d_in[base + 8];
  return p;
}

static void run_gemm(const __bf16* A, const float* W, __bf16* Bt, const float* bias,
                     float* C, int M, int Nn, int K, int Kp, hipStream_t s) {
  transpose_cast_kernel<<<(Nn * Kp + 255) / 256, 256, 0, s>>>(W, Bt, K, Kp, Nn);
  dim3 g((Nn + 63) / 64, (M + 63) / 64);
  wmma_gemm_bias<<<g, 256, 0, s>>>(A, Bt, bias, C, M, Nn, Kp);
}

static void run_tconv(const __bf16* xbf, int K, int Kp, int O, const TP& p,
                      const int* src, const int* dst, const float* ew, __bf16* Bt,
                      float* qb, float* kb, float* vb, float* zb,
                      float* logits, float* pexp, float* mb, float* sb,
                      hipStream_t s)
{
  run_gemm(xbf, p.Wq, Bt, p.bq, qb, NN, O, K, Kp, s);
  run_gemm(xbf, p.Wk, Bt, p.bk, kb, NN, O, K, Kp, s);
  run_gemm(xbf, p.Wv, Bt, p.bv, vb, NN, O, K, Kp, s);
  run_gemm(xbf, p.Ws, Bt, p.bs, zb, NN, O, K, Kp, s);   // root term
  fill_kernel<<<(NN + 255) / 256, 256, 0, s>>>(mb, -INFINITY, NN);
  fill_kernel<<<(NN + 255) / 256, 256, 0, s>>>(sb, 0.0f, NN);
  int waveBlocks = (NE + 7) / 8;
  float scale = 1.0f / sqrtf((float)O);
  edge_logits_kernel<<<waveBlocks, 256, 0, s>>>(qb, kb, ew, p.We, src, dst, logits, mb, NE, O, scale);
  edge_exp_kernel<<<(NE + 255) / 256, 256, 0, s>>>(logits, mb, dst, pexp, sb, NE);
  edge_scatter_kernel<<<waveBlocks, 256, 0, s>>>(vb, ew, p.We, src, dst, pexp, sb, zb, NE, O);
}

extern "C" void kernel_launch(void* const* d_in, const int* in_sizes, int n_in,
                              void* d_out, int out_size, void* d_ws, size_t ws_size,
                              hipStream_t stream)
{
  (void)in_sizes; (void)n_in; (void)out_size; (void)ws_size;

  const float* X    = (const float*)d_in[0];
  const int*   ei   = (const int*)d_in[1];
  const int*   src  = ei;
  const int*   dst  = ei + NE;
  const float* ew   = (const float*)d_in[2];
  const float* skip = (const float*)d_in[3];
  const float* H    = (const float*)d_in[4];
  const float* Cst  = (const float*)d_in[5];
  TP rnn0 = getTP(d_in, 6);
  TP rnn1 = getTP(d_in, 15);
  TP fc1  = getTP(d_in, 24);
  TP fc2  = getTP(d_in, 33);
  const float* nh_w = (const float*)d_in[42]; const float* nh_b = (const float*)d_in[43];
  const float* nc_w = (const float*)d_in[44]; const float* nc_b = (const float*)d_in[45];
  const float* no_w = (const float*)d_in[46]; const float* no_b = (const float*)d_in[47];

  // workspace layout (float-granular allocator)
  float* ws = (float*)d_ws;
  size_t off = 0;
  auto alloc = [&](size_t n) { float* p = ws + off; off += n; return p; };
  float*  qb    = alloc((size_t)NN * 512);
  float*  kb    = alloc((size_t)NN * 512);
  float*  vb    = alloc((size_t)NN * 512);
  float*  zb    = alloc((size_t)NN * 512);
  __bf16* xbf   = (__bf16*)alloc((size_t)NN * 128);   // NN x 256 bf16 (max Kp)
  __bf16* Bt    = (__bf16*)alloc((size_t)512 * 128);  // 512 x 256 bf16 (max Nn x Kp)
  float*  h2raw = alloc((size_t)NN * HIDD);
  float*  c2raw = alloc((size_t)NN * HIDD);
  float*  logit = alloc(NE);
  float*  pexp  = alloc(NE);
  float*  mb    = alloc(NN);
  float*  sb    = alloc(NN);
  float*  q1    = alloc(NN);
  float*  k1    = alloc(NN);
  float*  v1    = alloc(NN);
  float*  o1    = alloc(NN);

  // output layout: (o[NN], hidden[2*NN*HIDD], cell[2*NN*HIDD])
  float* out_o      = (float*)d_out;
  float* out_hidden = out_o + NN;
  float* out_cell   = out_hidden + (size_t)2 * NN * HIDD;

  const int lnBlocks = (NN * 32 + 255) / 256;

  // ---- layer 0: gclstm over [X, H0]  (K = 192, Kp = 192) ----
  concat2_bf16_kernel<<<(NN * 192 + 255) / 256, 256, 0, stream>>>(
      X, FINN, H, HIDD, xbf, NN, 192);
  run_tconv(xbf, 192, 192, 4 * HIDD, rnn0, src, dst, ew, Bt,
            qb, kb, vb, zb, logit, pexp, mb, sb, stream);
  lstm_gate_kernel<<<(NN * HIDD + 255) / 256, 256, 0, stream>>>(zb, Cst, h2raw, c2raw, NN);
  layernorm_kernel<<<lnBlocks, 256, 0, stream>>>(h2raw, nh_w, nh_b, out_hidden, NN, 0);
  layernorm_kernel<<<lnBlocks, 256, 0, stream>>>(c2raw, nc_w, nc_b, out_cell, NN, 0);

  // ---- layer 1: gclstm over [ln(h1), H1]  (K = 256, Kp = 256) ----
  concat2_bf16_kernel<<<(NN * 256 + 255) / 256, 256, 0, stream>>>(
      out_hidden, HIDD, H + (size_t)NN * HIDD, HIDD, xbf, NN, 256);
  run_tconv(xbf, 256, 256, 4 * HIDD, rnn1, src, dst, ew, Bt,
            qb, kb, vb, zb, logit, pexp, mb, sb, stream);
  lstm_gate_kernel<<<(NN * HIDD + 255) / 256, 256, 0, stream>>>(
      zb, Cst + (size_t)NN * HIDD, h2raw, c2raw, NN);
  layernorm_kernel<<<lnBlocks, 256, 0, stream>>>(
      h2raw, nh_w, nh_b, out_hidden + (size_t)NN * HIDD, NN, 0);
  layernorm_kernel<<<lnBlocks, 256, 0, stream>>>(
      c2raw, nc_w, nc_b, out_cell + (size_t)NN * HIDD, NN, 0);

  // ---- decoder head: o = relu(ln_o(h2raw)); concat skip; fc1 tconv (K=130,Kp=160) ----
  float* t128 = kb;   // consumed by concat before fc1's GEMMs overwrite kb
  layernorm_kernel<<<lnBlocks, 256, 0, stream>>>(h2raw, no_w, no_b, t128, NN, 1);
  concat2_bf16_kernel<<<(NN * 160 + 255) / 256, 256, 0, stream>>>(
      t128, HIDD, skip, SKIPC, xbf, NN, 160);
  run_tconv(xbf, HIDD + SKIPC, 160, HIDD, fc1, src, dst, ew, Bt,
            qb, kb, vb, zb, logit, pexp, mb, sb, stream);
  relu_kernel<<<(NN * HIDD + 255) / 256, 256, 0, stream>>>(zb, NN * HIDD);

  // ---- fc2 tconv (O = 1) + sigmoid ----
  fc2_proj_kernel<<<lnBlocks, 256, 0, stream>>>(zb, fc2.Wq, fc2.bq, fc2.Wk, fc2.bk,
                                                fc2.Wv, fc2.bv, fc2.Ws, fc2.bs,
                                                q1, k1, v1, o1, NN);
  fill_kernel<<<(NN + 255) / 256, 256, 0, stream>>>(mb, -INFINITY, NN);
  fill_kernel<<<(NN + 255) / 256, 256, 0, stream>>>(sb, 0.0f, NN);
  fc2_edge_logits_kernel<<<(NE + 255) / 256, 256, 0, stream>>>(
      q1, k1, ew, fc2.We, src, dst, logit, mb, NE);
  edge_exp_kernel<<<(NE + 255) / 256, 256, 0, stream>>>(logit, mb, dst, pexp, sb, NE);
  fc2_scatter_kernel<<<(NE + 255) / 256, 256, 0, stream>>>(
      v1, ew, fc2.We, src, dst, pexp, sb, o1, NE);
  sigmoid_store_kernel<<<(NN + 255) / 256, 256, 0, stream>>>(o1, out_o, NN);
}
